// MaxSigmoidAttnBlock_20779051778465
// MI455X (gfx1250) — matrix operand order repaired and checked by reading, not compiled, not run on hardware
//
#include <hip/hip_runtime.h>

typedef __attribute__((ext_vector_type(16))) _Float16 v16h;
typedef __attribute__((ext_vector_type(8)))  float    v8f;

// ---------------- geometry ----------------
#define BATCH   16
#define CIN     512
#define HW      40
#define NPIX    1600           // 40*40
#define PADW    42
#define PADHW   1764           // 42*42
#define EMB     256
#define COUT    512
#define NHEADS  8
#define NGUIDE  80
#define TOTCOL  (BATCH*NPIX)   // 25600

// workspace layout (bytes), all 256-aligned
#define OFF_XPAD  0ull                       // f16 [B][512][42][42]    28,901,376
#define OFF_WPSW  28901376ull                // f16 [144][512][32]       4,718,592
#define OFF_WESW  33619968ull                // f16 [16][256][32]          262,144
#define OFF_GSW   33882112ull                // f16 [128][80][32]          655,360
#define OFF_EPK   34537472ull                // f16 [128][1600][32]     13,107,200
#define OFF_ATTN  47644672ull                // f32 [128][1600]            819,200

// A-fragment K swizzle for 16-bit WMMA (lanes 0-15 read halves [0..7,16..23],
// lanes 16-31 read halves [8..15,24..31] as one contiguous 16-half run)
__device__ __forceinline__ int swz(int c) {
    const int base[4] = {0, 16, 8, 24};
    return base[c >> 3] + (c & 7);
}

// ---------------- pack kernels ----------------
__global__ void pack_xpad(const float* __restrict__ x, _Float16* __restrict__ xpad) {
    long idx = (long)blockIdx.x * blockDim.x + threadIdx.x;
    if (idx >= (long)BATCH * CIN * PADHW) return;
    int ww = idx % PADW;
    int hh = (idx / PADW) % PADW;
    long bc = idx / PADHW;                    // b*512+ci
    float v = 0.f;
    if (hh >= 1 && hh <= HW && ww >= 1 && ww <= HW)
        v = x[bc * NPIX + (hh - 1) * HW + (ww - 1)];
    xpad[idx] = (_Float16)v;
}

__global__ void pack_wp(const float* __restrict__ Wp, const float* __restrict__ gamma_p,
                        _Float16* __restrict__ Wpsw) {
    long idx = (long)blockIdx.x * blockDim.x + threadIdx.x;
    if (idx >= (long)COUT * CIN * 9) return;
    int t  = idx % 9;
    int ci = (idx / 9) % CIN;
    int co = idx / (9 * CIN);
    float scale = gamma_p[co] * rsqrtf(1.001f);   // BN: gamma/sqrt(1+1e-3)
    int kb = t * 16 + (ci >> 5);
    Wpsw[((long)kb * COUT + co) * 32 + swz(ci & 31)] = (_Float16)(Wp[idx] * scale);
}

__global__ void pack_we(const float* __restrict__ We, const float* __restrict__ gamma_e,
                        _Float16* __restrict__ Wesw) {
    long idx = (long)blockIdx.x * blockDim.x + threadIdx.x;
    if (idx >= (long)EMB * CIN) return;
    int ci = idx % CIN;
    int e  = idx / CIN;
    float scale = gamma_e[e] * rsqrtf(1.001f);
    int kb = ci >> 5;
    Wesw[((long)kb * EMB + e) * 32 + swz(ci & 31)] = (_Float16)(We[idx] * scale);
}

// guide_fc: g[b,n,e] = guide[b,n,:]·Wg[e,:] + bg[e], stored A-frag-swizzled per (b,head)
__global__ void guide_fc(const float* __restrict__ guide, const float* __restrict__ Wg,
                         const float* __restrict__ bg, _Float16* __restrict__ g_sw) {
    int idx = blockIdx.x * blockDim.x + threadIdx.x;
    if (idx >= BATCH * NGUIDE * EMB) return;
    int e = idx & 255;
    int n = (idx >> 8) % NGUIDE;
    int b = idx / (NGUIDE * EMB);
    const float* gr = guide + ((long)(b * NGUIDE + n)) * CIN;
    const float* wr = Wg + (long)e * CIN;
    float s = bg[e];
    #pragma unroll 4
    for (int c = 0; c < CIN; ++c) s += gr[c] * wr[c];
    int m = e >> 5, cc = e & 31;
    g_sw[(((long)(b * NHEADS + m) * NGUIDE + n) << 5) + swz(cc)] = (_Float16)s;
}

// ---------------- fused implicit-GEMM conv (embed 1x1 / project 3x3) ----------------
// 256 threads = 8 waves (4 along M x 2 along N); 128x128 tile, K-block 32.
template<int KBLOCKS, int WROWS, bool EMBED>
__global__ __launch_bounds__(256) void conv_gemm(
    const _Float16* __restrict__ Wsw,     // [kb][WROWS][32], A-frag swizzled, BN-gamma folded
    const _Float16* __restrict__ xpad,    // [B][512][42][42]
    const float*    __restrict__ beta,    // beta_e / beta_p
    const float*    __restrict__ attnbuf, // project only
    _Float16*       __restrict__ e_pk,    // embed only: [b*8+m][pix][32]
    float*          __restrict__ out)     // project only: [b][512][40][40]
{
    __shared__ _Float16 Asmem[128 * 32];
    __shared__ _Float16 Bsmem[128 * 32];

    const int tid  = threadIdx.x;
    const int lane = tid & 31;
    const int wid  = tid >> 5;
    const int wm   = wid & 3;            // 0..3 -> 32-row stripe
    const int wn   = wid >> 2;           // 0..1 -> 64-col stripe
    const int l15  = lane & 15;
    const int hi   = (lane >> 4) << 4;   // 0 or 16 halves

    const int col0 = blockIdx.x * 128;
    const int co0  = blockIdx.y * 128;

    // fixed B-stage column per thread
    const int nB    = tid & 127;
    const int khalf = tid >> 7;          // even/odd k stripes
    const int colB  = col0 + nB;
    const int bB    = colB / NPIX;
    const int pixB  = colB - bB * NPIX;
    const int hB    = pixB / HW, wB = pixB - hB * HW;
    const long baseB = (long)bB * CIN * PADHW + hB * PADW + wB;

    v8f acc[2][4];
    #pragma unroll
    for (int i = 0; i < 2; ++i)
        #pragma unroll
        for (int j = 0; j < 4; ++j)
            #pragma unroll
            for (int e = 0; e < 8; ++e) acc[i][j][e] = 0.f;

    const unsigned ldsA = (unsigned)(unsigned long long)(const void*)&Asmem[0];

    for (int kb = 0; kb < KBLOCKS; ++kb) {
        const int t   = EMBED ? 4 : (kb >> 4);
        const int cb  = EMBED ? (kb << 5) : ((kb & 15) << 5);
        const int dyx = (t / 3) * PADW + (t % 3);

        // ---- A tile: 8KB contiguous, async DMA to LDS (TDM-style path) ----
        {
            unsigned long long ga = (unsigned long long)(Wsw + ((long)kb * WROWS + co0) * 32)
                                    + (unsigned long long)tid * 16ull;
            unsigned la = ldsA + (unsigned)tid * 16u;
            asm volatile("global_load_async_to_lds_b128 %0, %1, off"
                         :: "v"(la), "v"(ga) : "memory");
            asm volatile("global_load_async_to_lds_b128 %0, %1, off"
                         :: "v"(la + 4096u), "v"(ga + 4096ull) : "memory");
        }

        // ---- B tile: gather x patch, k-contiguous per column ----
        {
            const _Float16* xb = xpad + baseB + (long)cb * PADHW + dyx;
            #pragma unroll
            for (int i = 0; i < 16; ++i) {
                int k = 2 * i + khalf;
                Bsmem[nB * 32 + swz(k)] = xb[(long)k * PADHW];
            }
        }
        asm volatile("s_wait_asynccnt 0" ::: "memory");
        __syncthreads();

        // ---- 8 x v_wmma_f32_16x16x32_f16 per wave ----
        v16h a0 = *(const v16h*)&Asmem[(wm * 32 + l15) * 32 + hi];
        v16h a1 = *(const v16h*)&Asmem[(wm * 32 + 16 + l15) * 32 + hi];
        #pragma unroll
        for (int j = 0; j < 4; ++j) {
            v16h bf = *(const v16h*)&Bsmem[(wn * 64 + j * 16 + l15) * 32 + hi];
            acc[0][j] = __builtin_amdgcn_wmma_f32_16x16x32_f16(false, a0, false, bf,
                          (short)0, acc[0][j], false, false);
            acc[1][j] = __builtin_amdgcn_wmma_f32_16x16x32_f16(false, a1, false, bf,
                          (short)0, acc[1][j], false, false);
        }
        __syncthreads();
    }

    // ---- epilogue ----
    #pragma unroll
    for (int i = 0; i < 2; ++i) {
        #pragma unroll
        for (int j = 0; j < 4; ++j) {
            const int coB = co0 + wm * 32 + i * 16 + (hi >> 1);  // +8 for upper lanes
            const int col = col0 + wn * 64 + j * 16 + l15;
            const int b   = col / NPIX;
            const int pix = col - b * NPIX;
            #pragma unroll
            for (int e = 0; e < 8; ++e) {
                const int co = coB + e;
                float v = acc[i][j][e] + beta[co];
                if (EMBED) {
                    e_pk[(((long)(b * NHEADS + (co >> 5)) * NPIX + pix) << 5) + swz(co & 31)] =
                        (_Float16)v;
                } else {
                    float a = attnbuf[(long)(b * NHEADS + (co >> 6)) * NPIX + pix];
                    out[((long)(b * COUT + co)) * NPIX + pix] = v * a;
                }
            }
        }
    }
}

// ---------------- attention: per (b,m), max_n sigmoid(g·e) ----------------
__global__ __launch_bounds__(32) void attn_kernel(
    const _Float16* __restrict__ g_sw,   // [bm][80][32] swizzled
    const _Float16* __restrict__ e_pk,   // [bm][1600][32]
    const float*    __restrict__ head_bias,
    float*          __restrict__ attnbuf)
{
    const int lane = threadIdx.x;
    const int l15  = lane & 15;
    const int hi   = (lane >> 4) << 4;
    const int bm   = blockIdx.y;             // b*8+m
    const int m    = bm & 7;
    const int col0 = blockIdx.x * 64;

    v16h bfr[4];
    #pragma unroll
    for (int j = 0; j < 4; ++j)
        bfr[j] = *(const v16h*)(e_pk + (((long)bm * NPIX + col0 + j * 16 + l15) << 5) + hi);

    v8f cmax[4];
    #pragma unroll
    for (int j = 0; j < 4; ++j)
        #pragma unroll
        for (int e = 0; e < 8; ++e) cmax[j][e] = -3.0e38f;

    for (int mt = 0; mt < 5; ++mt) {         // 80 guide tokens = 5 M-tiles
        v16h a = *(const v16h*)(g_sw + (((long)bm * NGUIDE + mt * 16 + l15) << 5) + hi);
        #pragma unroll
        for (int j = 0; j < 4; ++j) {
            v8f z;
            #pragma unroll
            for (int e = 0; e < 8; ++e) z[e] = 0.f;
            z = __builtin_amdgcn_wmma_f32_16x16x32_f16(false, a, false, bfr[j],
                  (short)0, z, false, false);
            #pragma unroll
            for (int e = 0; e < 8; ++e) cmax[j][e] = fmaxf(cmax[j][e], z[e]);
        }
    }

    const float hb = head_bias[m];
    #pragma unroll
    for (int j = 0; j < 4; ++j) {
        float s = cmax[j][0];
        #pragma unroll
        for (int e = 1; e < 8; ++e) s = fmaxf(s, cmax[j][e]);
        s = fmaxf(s, __shfl_xor(s, 16, 32));           // combine M 0-7 with 8-15
        float atn = 1.f / (1.f + __expf(-(s * 0.17677669529663689f + hb)));
        if (lane < 16)
            attnbuf[(long)bm * NPIX + col0 + j * 16 + lane] = atn;
    }
}

// ---------------- launch ----------------
extern "C" void kernel_launch(void* const* d_in, const int* in_sizes, int n_in,
                              void* d_out, int out_size, void* d_ws, size_t ws_size,
                              hipStream_t stream) {
    const float* x         = (const float*)d_in[0];
    const float* guide     = (const float*)d_in[1];
    const float* We        = (const float*)d_in[2];
    const float* gamma_e   = (const float*)d_in[3];
    const float* beta_e    = (const float*)d_in[4];
    const float* Wg        = (const float*)d_in[5];
    const float* bg        = (const float*)d_in[6];
    const float* head_bias = (const float*)d_in[7];
    const float* Wp        = (const float*)d_in[8];
    const float* gamma_p   = (const float*)d_in[9];
    const float* beta_p    = (const float*)d_in[10];
    float* out = (float*)d_out;

    char* ws = (char*)d_ws;
    _Float16* xpad = (_Float16*)(ws + OFF_XPAD);
    _Float16* Wpsw = (_Float16*)(ws + OFF_WPSW);
    _Float16* Wesw = (_Float16*)(ws + OFF_WESW);
    _Float16* g_sw = (_Float16*)(ws + OFF_GSW);
    _Float16* e_pk = (_Float16*)(ws + OFF_EPK);
    float*    attn = (float*)(ws + OFF_ATTN);

    {   // pack x (padded, f16)
        long n = (long)BATCH * CIN * PADHW;
        pack_xpad<<<dim3((n + 255) / 256), dim3(256), 0, stream>>>(x, xpad);
    }
    {   // pack weights (BN gamma folded, A-frag swizzled)
        long n = (long)COUT * CIN * 9;
        pack_wp<<<dim3((n + 255) / 256), dim3(256), 0, stream>>>(Wp, gamma_p, Wpsw);
        long n2 = (long)EMB * CIN;
        pack_we<<<dim3((n2 + 255) / 256), dim3(256), 0, stream>>>(We, gamma_e, Wesw);
    }
    {   // guide fc
        int n = BATCH * NGUIDE * EMB;
        guide_fc<<<dim3((n + 255) / 256), dim3(256), 0, stream>>>(guide, Wg, bg, g_sw);
    }
    // embed 1x1 conv + BN -> e_pk (f16, per-head packed)
    conv_gemm<16, EMB, true><<<dim3(TOTCOL / 128, EMB / 128), dim3(256), 0, stream>>>(
        Wesw, xpad, beta_e, nullptr, e_pk, nullptr);
    // attention gate
    attn_kernel<<<dim3(NPIX / 64, BATCH * NHEADS), dim3(32), 0, stream>>>(
        g_sw, e_pk, head_bias, attn);
    // project 3x3 conv + BN + gate -> out
    conv_gemm<144, COUT, false><<<dim3(TOTCOL / 128, COUT / 128), dim3(256), 0, stream>>>(
        Wpsw, xpad, beta_p, attn, nullptr, out);
}
